// Attention_44066364457602
// MI455X (gfx1250) — matrix operand order repaired and checked by reading, not compiled, not run on hardware
//
#include <hip/hip_runtime.h>
#include <hip/hip_bf16.h>
#include <stdint.h>

// ---------------------------------------------------------------------------
// MHA forward for MI455X (gfx1250), wave32 + WMMA bf16 (f32 accumulate).
// Stage 0 converts all operands to bf16 once; the GEMM/attention kernels run
// pure vector-load + v_wmma with no per-element conversion VALU.
// ---------------------------------------------------------------------------

typedef __attribute__((ext_vector_type(16))) __bf16 v16bf;
typedef __attribute__((ext_vector_type(8)))  __bf16 v8bf;
typedef __attribute__((ext_vector_type(8)))  float  v8f;

#define WMMA_BF16(a, b, c) \
  __builtin_amdgcn_wmma_f32_16x16x32_bf16(false, (a), false, (b), (short)0, (c), false, false)

constexpr int Bc  = 2;
constexpr int Sc  = 2048;
constexpr int Dc  = 1024;
constexpr int Hc  = 16;
constexpr int DHc = 64;
constexpr float ATTN_NEG = -1.0e30f;

__device__ __forceinline__ __bf16 f2bf(float f) {
  union { float f; uint32_t u; } c;
  c.f = f;
  uint32_t r = c.u + 0x7FFFu + ((c.u >> 16) & 1u);   // round-to-nearest-even
  uint16_t h = (uint16_t)(r >> 16);
  return __builtin_bit_cast(__bf16, h);
}

// ---------------------------------------------------------------------------
// Kernel 0a: elementwise f32 -> bf16 (x, W_Q, W_K, W_V). Memory bound, runs once.
// ---------------------------------------------------------------------------
__global__ __launch_bounds__(256)
void cvt_bf16_kernel(const float* __restrict__ src, __bf16* __restrict__ dst, int n) {
  for (int i = blockIdx.x * 256 + threadIdx.x; i < n; i += gridDim.x * 256)
    dst[i] = f2bf(src[i]);
}

// ---------------------------------------------------------------------------
// Kernel 0b: W_O [H, D, DH] f32 -> Wo2 [D, H*DH] bf16 (k-major for B fragments).
// ---------------------------------------------------------------------------
__global__ __launch_bounds__(256)
void cvt_wo_kernel(const float* __restrict__ src, __bf16* __restrict__ dst) {
  const int n = Hc * Dc * DHc;
  for (int i = blockIdx.x * 256 + threadIdx.x; i < n; i += gridDim.x * 256) {
    const int h   = i / (Dc * DHc);
    const int rem = i - h * (Dc * DHc);
    const int d   = rem / DHc;
    const int dh  = rem - d * DHc;
    dst[(size_t)d * (Hc * DHc) + h * DHc + dh] = f2bf(src[i]);
  }
}

// ---------------------------------------------------------------------------
// Kernel 1: fused QKV projection (all-bf16 operands).
// Grid: B * H * (S/128), 256 threads (8 waves). Wave: 16 rows x 64 cols of
// Q, K, V. One A fragment feeds 12 WMMAs per K-step.
// Q, K stored [B,H,S,DH]; V stored transposed [B,H,DH,S] for stage 2.
// ---------------------------------------------------------------------------
__global__ __launch_bounds__(256)
void qkv_proj_kernel(const __bf16* __restrict__ x,
                     const __bf16* __restrict__ Wq,
                     const __bf16* __restrict__ Wk,
                     const __bf16* __restrict__ Wv,
                     const float* __restrict__ bq,
                     const float* __restrict__ bk,
                     const float* __restrict__ bv,
                     __bf16* __restrict__ Qo,
                     __bf16* __restrict__ Ko,
                     __bf16* __restrict__ Vt)
{
  const int lane = threadIdx.x & 31;
  const int wave = threadIdx.x >> 5;
  const int half = lane >> 4;
  const int ln16 = lane & 15;

  int bid = blockIdx.x;
  const int rowTiles = Sc / 128;            // 16
  const int rt = bid % rowTiles;  bid /= rowTiles;
  const int h  = bid % Hc;        bid /= Hc;
  const int b  = bid;

  const int rowBase = rt * 128 + wave * 16;
  const __bf16* __restrict__ xb = x + (size_t)b * Sc * Dc;

  v8f accq[4] = {}, acck[4] = {}, accv[4] = {};

  for (int k0 = 0; k0 < Dc; k0 += 32) {
    // --- A fragment (16x32 of x). Lane = row (ln16); half selects K bank.
    v16bf a;
    {
      const __bf16* xr = xb + (size_t)(rowBase + ln16) * Dc + k0 + half * 16;
      __builtin_prefetch(xr + 32, 0, 0);
      v8bf lo = *(const v8bf*)(xr);         // K = half*16 + 0..7
      v8bf hi = *(const v8bf*)(xr + 8);     // K = half*16 + 8..15
      #pragma unroll
      for (int j = 0; j < 8; ++j) { a[j] = lo[j]; a[8 + j] = hi[j]; }
    }
    #pragma unroll
    for (int t = 0; t < 4; ++t) {
      const int ncol = t * 16 + ln16;
      const size_t wbase = ((size_t)h * DHc + ncol) * Dc + k0 + half * 16;
      // B element j: K = half*16 + j  -> one contiguous 32B vector each
      v16bf fq = *(const v16bf*)(Wq + wbase);
      v16bf fk = *(const v16bf*)(Wk + wbase);
      v16bf fv = *(const v16bf*)(Wv + wbase);
      accq[t] = WMMA_BF16(a, fq, accq[t]);
      acck[t] = WMMA_BF16(a, fk, acck[t]);
      accv[t] = WMMA_BF16(a, fv, accv[t]);
    }
  }

  // Store (bias add, convert to bf16).
  const size_t plane = ((size_t)b * Hc + h) * Sc;
  #pragma unroll
  for (int t = 0; t < 4; ++t) {
    const int col = t * 16 + ln16;
    const float biq = bq[h * DHc + col];
    const float bik = bk[h * DHc + col];
    const float biv = bv[h * DHc + col];
    // Q, K row-major [S, DH]
    #pragma unroll
    for (int v = 0; v < 8; ++v) {
      const int row = rowBase + v + half * 8;
      const size_t idx = (plane + row) * DHc + col;
      Qo[idx] = f2bf(accq[t][v] + biq);
      Ko[idx] = f2bf(acck[t][v] + bik);
    }
    // V transposed [DH, S]: rows v are consecutive -> one 16B store
    v8bf vv;
    #pragma unroll
    for (int v = 0; v < 8; ++v) vv[v] = f2bf(accv[t][v] + biv);
    const size_t vidx = (((size_t)b * Hc + h) * DHc + col) * Sc + rowBase + half * 8;
    *(v8bf*)(Vt + vidx) = vv;
  }
}

// ---------------------------------------------------------------------------
// Kernel 2: causal flash attention, all matmuls in bf16 WMMA.
// Grid: B * H * (S/128), 256 threads. Wave = 16-row q tile; 32-wide kv tiles
// with online softmax. P relaid out C->A via per-wave bf16 LDS tile.
// V consumed transposed -> contiguous 32B B-fragment loads.
// Z written [B*S, H*DH] bf16.
// ---------------------------------------------------------------------------
__global__ __launch_bounds__(256)
void flash_attn_kernel(const __bf16* __restrict__ Q,
                       const __bf16* __restrict__ K,
                       const __bf16* __restrict__ Vt,
                       __bf16* __restrict__ Z)
{
  __shared__ __bf16 plds[8][16][32];   // per-wave P tile (8 KB)

  const int lane = threadIdx.x & 31;
  const int wave = threadIdx.x >> 5;
  const int half = lane >> 4;
  const int ln16 = lane & 15;

  int bid = blockIdx.x;
  const int rowTiles = Sc / 128;
  const int rt = bid % rowTiles;  bid /= rowTiles;
  const int h  = bid % Hc;        bid /= Hc;
  const int b  = bid;

  const int qBase = rt * 128 + wave * 16;
  const size_t plane = ((size_t)b * Hc + h) * Sc;
  const __bf16* __restrict__ Qp = Q + plane * DHc;
  const __bf16* __restrict__ Kp = K + plane * DHc;
  const __bf16* __restrict__ Vp = Vt + plane * DHc;   // [DH, S] per (b,h)

  // --- Q A-fragments (two 32-wide K slices of DH), loaded once.
  v16bf aq0, aq1;
  {
    const __bf16* qr = Qp + (size_t)(qBase + ln16) * DHc;
    #pragma unroll
    for (int c = 0; c < 2; ++c) {
      v8bf lo = *(const v8bf*)(qr + c * 32 + half * 16);
      v8bf hi = *(const v8bf*)(qr + c * 32 + half * 16 + 8);
      #pragma unroll
      for (int j = 0; j < 8; ++j) {
        if (c == 0) { aq0[j] = lo[j]; aq0[8 + j] = hi[j]; }
        else        { aq1[j] = lo[j]; aq1[8 + j] = hi[j]; }
      }
    }
  }

  float mrun[8], lrun[8];
  #pragma unroll
  for (int v = 0; v < 8; ++v) { mrun[v] = ATTN_NEG; lrun[v] = 0.f; }
  v8f o[4] = {};

  const float sc = 0.125f;                    // 1/sqrt(DH)
  const int ktMax = (qBase + 15) / 32;        // causal upper bound

  for (int kt = 0; kt <= ktMax; ++kt) {
    const int kvBase = kt * 32;

    // --- scores S = Q * K^T (two 16x16 C fragments covering 32 kv cols)
    v8f s0 = {}, s1 = {};
    {
      const __bf16* kr0 = Kp + (size_t)(kvBase + ln16) * DHc + half * 16;
      v16bf b0 = *(const v16bf*)(kr0);        // dh 0..31 slice
      v16bf b1 = *(const v16bf*)(kr0 + 32);   // dh 32..63 slice
      s0 = WMMA_BF16(aq0, b0, s0);
      s0 = WMMA_BF16(aq1, b1, s0);
      const __bf16* kr1 = Kp + (size_t)(kvBase + 16 + ln16) * DHc + half * 16;
      v16bf b2 = *(const v16bf*)(kr1);
      v16bf b3 = *(const v16bf*)(kr1 + 32);
      s1 = WMMA_BF16(aq0, b2, s1);
      s1 = WMMA_BF16(aq1, b3, s1);
    }

    // --- causal mask + scale
    float t0[8], t1[8];
    #pragma unroll
    for (int v = 0; v < 8; ++v) {
      const int qrow = qBase + v + half * 8;
      const int c0 = kvBase + ln16;
      t0[v] = (c0      <= qrow) ? s0[v] * sc : ATTN_NEG;
      t1[v] = (c0 + 16 <= qrow) ? s1[v] * sc : ATTN_NEG;
    }

    // --- row max across 16 lanes of this half (covers 32 kv cols)
    float rmax[8];
    #pragma unroll
    for (int v = 0; v < 8; ++v) rmax[v] = fmaxf(t0[v], t1[v]);
    #pragma unroll
    for (int m = 1; m < 16; m <<= 1) {
      #pragma unroll
      for (int v = 0; v < 8; ++v)
        rmax[v] = fmaxf(rmax[v], __shfl_xor(rmax[v], m, 32));
    }

    float alpha[8], mnew[8];
    #pragma unroll
    for (int v = 0; v < 8; ++v) {
      mnew[v]  = fmaxf(mrun[v], rmax[v]);
      alpha[v] = __expf(mrun[v] - mnew[v]);
      mrun[v]  = mnew[v];
    }

    float p0[8], p1[8], rsum[8];
    #pragma unroll
    for (int v = 0; v < 8; ++v) {
      p0[v] = __expf(t0[v] - mnew[v]);
      p1[v] = __expf(t1[v] - mnew[v]);
      rsum[v] = p0[v] + p1[v];
    }
    #pragma unroll
    for (int m = 1; m < 16; m <<= 1) {
      #pragma unroll
      for (int v = 0; v < 8; ++v)
        rsum[v] += __shfl_xor(rsum[v], m, 32);
    }
    #pragma unroll
    for (int v = 0; v < 8; ++v) lrun[v] = lrun[v] * alpha[v] + rsum[v];

    // rescale accumulators
    #pragma unroll
    for (int t = 0; t < 4; ++t)
      #pragma unroll
      for (int v = 0; v < 8; ++v) o[t][v] *= alpha[v];

    // --- relayout P: C fragment -> bf16 LDS -> A fragment
    #pragma unroll
    for (int v = 0; v < 8; ++v) {
      plds[wave][v + half * 8][ln16]      = f2bf(p0[v]);
      plds[wave][v + half * 8][ln16 + 16] = f2bf(p1[v]);
    }
    __builtin_amdgcn_wave_barrier();
    v16bf pa;
    {
      // element j: k = half*16 + j -> two 8-element runs
      v8bf lo = *(const v8bf*)(&plds[wave][ln16][half * 16]);
      v8bf hi = *(const v8bf*)(&plds[wave][ln16][half * 16 + 8]);
      #pragma unroll
      for (int j = 0; j < 8; ++j) { pa[j] = lo[j]; pa[8 + j] = hi[j]; }
    }
    __builtin_amdgcn_wave_barrier();

    // --- O += P * V   (Vt: lane n = dh row; element j: kv = half*16+j, contiguous)
    #pragma unroll
    for (int t = 0; t < 4; ++t) {
      const __bf16* vr = Vp + (size_t)(t * 16 + ln16) * Sc + kvBase + half * 16;
      v16bf bv = *(const v16bf*)(vr);
      o[t] = WMMA_BF16(pa, bv, o[t]);
    }
  }

  // --- finalize: divide by denominator, store Z bf16 [B*S, H*DH]
  float inv[8];
  #pragma unroll
  for (int v = 0; v < 8; ++v) inv[v] = 1.0f / lrun[v];
  #pragma unroll
  for (int t = 0; t < 4; ++t) {
    #pragma unroll
    for (int v = 0; v < 8; ++v) {
      const int qrow = qBase + v + half * 8;
      const size_t idx = ((size_t)b * Sc + qrow) * (Hc * DHc) + h * DHc + t * 16 + ln16;
      Z[idx] = f2bf(o[t][v] * inv[v]);
    }
  }
}

// ---------------------------------------------------------------------------
// Kernel 3: output projection. out[r, d] = sum_k Z[r, k] * Wo2[d, k] + bo[d].
// Grid: (B*S/128) * (D/64), 256 threads; wave = 16 rows x 64 cols.
// ---------------------------------------------------------------------------
__global__ __launch_bounds__(256)
void out_proj_kernel(const __bf16* __restrict__ Z,
                     const __bf16* __restrict__ Wo2,
                     const float* __restrict__ bo,
                     float* __restrict__ out)
{
  const int lane = threadIdx.x & 31;
  const int wave = threadIdx.x >> 5;
  const int half = lane >> 4;
  const int ln16 = lane & 15;

  int bid = blockIdx.x;
  const int colTiles = Dc / 64;               // 16
  const int ct = bid % colTiles;
  const int rtile = bid / colTiles;

  const int rowBase = rtile * 128 + wave * 16;
  const int colBase = ct * 64;

  v8f acc[4] = {};

  for (int k0 = 0; k0 < Hc * DHc; k0 += 32) {
    v16bf a;
    {
      const __bf16* zr = Z + (size_t)(rowBase + ln16) * (Hc * DHc) + k0 + half * 16;
      __builtin_prefetch(zr + 32, 0, 0);
      v8bf lo = *(const v8bf*)(zr);
      v8bf hi = *(const v8bf*)(zr + 8);
      #pragma unroll
      for (int j = 0; j < 8; ++j) { a[j] = lo[j]; a[8 + j] = hi[j]; }
    }
    const int kb = k0 + half * 16;            // B element j: k = kb + j
    #pragma unroll
    for (int t = 0; t < 4; ++t) {
      const int col = colBase + t * 16 + ln16;
      v16bf bw = *(const v16bf*)(Wo2 + (size_t)col * (Hc * DHc) + kb);
      acc[t] = WMMA_BF16(a, bw, acc[t]);
    }
  }

  #pragma unroll
  for (int t = 0; t < 4; ++t) {
    const int col = colBase + t * 16 + ln16;
    const float bias = bo[col];
    #pragma unroll
    for (int v = 0; v < 8; ++v) {
      const int row = rowBase + v + half * 8;
      out[(size_t)row * Dc + col] = acc[t][v] + bias;
    }
  }
}

// ---------------------------------------------------------------------------
// Host-side launch.
// Inputs (setup_inputs order): x, W_Q, W_K, W_V, W_O, b_Q, b_K, b_V, b_O
// Workspace (bf16): xw | Wq | Wk | Wv | Wo2 | Q | K | Vt | Z  (~48 MB)
// ---------------------------------------------------------------------------
extern "C" void kernel_launch(void* const* d_in, const int* in_sizes, int n_in,
                              void* d_out, int out_size, void* d_ws, size_t ws_size,
                              hipStream_t stream) {
  (void)in_sizes; (void)n_in; (void)out_size; (void)ws_size;

  const float* x   = (const float*)d_in[0];
  const float* W_Q = (const float*)d_in[1];
  const float* W_K = (const float*)d_in[2];
  const float* W_V = (const float*)d_in[3];
  const float* W_O = (const float*)d_in[4];
  const float* b_Q = (const float*)d_in[5];
  const float* b_K = (const float*)d_in[6];
  const float* b_V = (const float*)d_in[7];
  const float* b_O = (const float*)d_in[8];
  float* out = (float*)d_out;

  const size_t XSZ = (size_t)Bc * Sc * Dc;         // 4,194,304
  const size_t WSZ = (size_t)Hc * DHc * Dc;        // 1,048,576
  const size_t QSZ = (size_t)Bc * Hc * Sc * DHc;   // 4,194,304

  __bf16* xw  = (__bf16*)d_ws;
  __bf16* Wqw = xw  + XSZ;
  __bf16* Wkw = Wqw + WSZ;
  __bf16* Wvw = Wkw + WSZ;
  __bf16* Wow = Wvw + WSZ;   // [D, H*DH]
  __bf16* Qw  = Wow + WSZ;
  __bf16* Kw  = Qw  + QSZ;
  __bf16* Vw  = Kw  + QSZ;   // transposed [B,H,DH,S]
  __bf16* Zw  = Vw  + QSZ;

  const dim3 block(256);

  // Stage 0: one-shot conversions (memory bound).
  cvt_bf16_kernel<<<dim3(2048), block, 0, stream>>>(x,   xw,  (int)XSZ);
  cvt_bf16_kernel<<<dim3(1024), block, 0, stream>>>(W_Q, Wqw, (int)WSZ);
  cvt_bf16_kernel<<<dim3(1024), block, 0, stream>>>(W_K, Wkw, (int)WSZ);
  cvt_bf16_kernel<<<dim3(1024), block, 0, stream>>>(W_V, Wvw, (int)WSZ);
  cvt_wo_kernel  <<<dim3(1024), block, 0, stream>>>(W_O, Wow);

  const dim3 gridQKV(Bc * Hc * (Sc / 128));        // 512
  const dim3 gridAttn(Bc * Hc * (Sc / 128));       // 512
  const dim3 gridOut((Bc * Sc / 128) * (Dc / 64)); // 512

  qkv_proj_kernel<<<gridQKV, block, 0, stream>>>(xw, Wqw, Wkw, Wvw,
                                                 b_Q, b_K, b_V, Qw, Kw, Vw);
  flash_attn_kernel<<<gridAttn, block, 0, stream>>>(Qw, Kw, Vw, Zw);
  out_proj_kernel<<<gridOut, block, 0, stream>>>(Zw, W_O ? Wow : Wow, b_O, out);
}
